// SpikingHybridCoreFlow_87359634800665
// MI455X (gfx1250) — matrix operand order repaired and checked by reading, not compiled, not run on hardware
//
#include <hip/hip_runtime.h>

// ---------------------------------------------------------------------------
// SpikingHybridCoreFlow on MI455X (gfx1250)
//   per step, per core c:  memb[c](512x16) += W[c](512x512) x signals(512x16)
//   signals are binary 0/1 -> exact bf16; W split into bf16 hi+lo so the two
//   v_wmma_f32_16x16x32_bf16 per tile approximate the fp32 product to ~16 bits.
//   Each wave owns TWO M-tiles (2 independent accumulator chains); the K-loop
//   is explicitly software-pipelined: loads for K-step k+1 are issued into a
//   second register set before the 4 WMMAs of step k, so global/ds loads
//   overlap the matrix pipe instead of draining loadcnt to 0 per WMMA.
// ---------------------------------------------------------------------------

typedef __attribute__((ext_vector_type(16))) __bf16 v16bf;
typedef __attribute__((ext_vector_type(8)))  float  v8f;

#define C_   64
#define A_   512
#define N_   512
#define B_   16
#define IN_  8192
#define OUT_ 1024
#define T_   32

#define WMMA_BF16(A, Bf, Cc) \
  __builtin_amdgcn_wmma_f32_16x16x32_bf16(false, (A), false, (Bf), (short)0, (Cc), false, false)

// ---------------------------------------------------------------------------
// Pre-swizzle W (fp32, [C][N][A]) into WMMA A-operand layout, split hi/lo bf16.
// Storage index s = ((c*32 + mt)*16 + kt)*512 + lane*16 + j
//   lane 0-15 : row = lane,     K = j
//   lane16-31 : row = lane-16,  K = j+16
// (contiguous K order, not the ISA table order; the B fragment from sig uses the same)
// ---------------------------------------------------------------------------
__global__ __launch_bounds__(256) void prep_w_kernel(const float* __restrict__ W,
                                                     __bf16* __restrict__ Whi,
                                                     __bf16* __restrict__ Wlo)
{
  unsigned s    = blockIdx.x * 256u + threadIdx.x;   // 0 .. C*N*A-1
  unsigned j    = s & 15u;
  unsigned lane = (s >> 4) & 31u;
  unsigned kt   = (s >> 9) & 15u;
  unsigned mt   = (s >> 13) & 31u;
  unsigned c    = s >> 18;
  unsigned row  = lane & 15u;
  unsigned K    = (lane < 16u) ? j
                               : j + 16u;
  unsigned n = mt * 16u + row;
  unsigned a = kt * 32u + K;
  float w  = W[((size_t)c * N_ + n) * A_ + a];
  __bf16 hi = (__bf16)w;
  float  lo = w - (float)hi;
  Whi[s] = hi;
  Wlo[s] = (__bf16)lo;
}

// ---------------------------------------------------------------------------
// Zero membrane, both fired-buffers, and the output accumulator.
// ---------------------------------------------------------------------------
__global__ __launch_bounds__(256) void zero_state_kernel(float* __restrict__ memb,
                                                         __bf16* __restrict__ buf0,
                                                         __bf16* __restrict__ buf1,
                                                         float* __restrict__ out)
{
  unsigned i = blockIdx.x * 256u + threadIdx.x;      // 0 .. C*B*N-1
  memb[i] = 0.0f;
  buf0[i] = (__bf16)0.0f;
  buf1[i] = (__bf16)0.0f;
  if (i < (unsigned)(B_ * OUT_)) out[i] = 0.0f;
}

// uniform spike train (matches reference _uniform_spikes for cycle < T)
__device__ __forceinline__ float spike_val(float xv, float cyc)
{
  float Nspk = rintf(xv * 32.0f);                    // jnp.round -> rint
  if (Nspk >= 32.0f) return 1.0f;
  if (Nspk <= 0.0f)  return 0.0f;
  float spacing = 32.0f / Nspk;
  bool res = (floorf(cyc / spacing) < Nspk) &&
             (floorf(fmodf(cyc, spacing)) == 0.0f);
  return res ? 1.0f : 0.0f;
}

// ---------------------------------------------------------------------------
// One timestep. grid = (C cores, 2 M-splits), block = 256 threads = 8 waves.
// Wave w handles M-tiles mt0 = ms*16 + w and mt1 = mt0 + 8.
// ---------------------------------------------------------------------------
__global__ __launch_bounds__(256) void step_kernel(
    const float*  __restrict__ x,
    const __bf16* __restrict__ Whi,
    const __bf16* __restrict__ Wlo,
    const float*  __restrict__ thresholds,
    const int*    __restrict__ src_core,
    const int*    __restrict__ src_index,
    float*        __restrict__ memb,
    const __bf16* __restrict__ buf_prev,
    __bf16*       __restrict__ buf_next,
    int cycle)
{
  __shared__ __align__(64) __bf16 sig[B_ * A_];      // sig[b][a], 16 KB
  const int c   = blockIdx.x;
  const int ms  = blockIdx.y;                        // 0..1
  const int tid = threadIdx.x;
  const float cyc = (float)cycle;

  // ---- gather binary signal matrix for this core into LDS (b-major) ----
  for (int i = tid; i < B_ * A_; i += 256) {
    int a  = i >> 4;
    int b  = i & 15;
    int sc = src_core[c * A_ + a];
    int si = src_index[c * A_ + a];
    float v;
    if (sc < 0) {
      v = spike_val(x[b * IN_ + si], cyc);
    } else {
      v = (float)buf_prev[(size_t)b * (C_ * N_) + sc * N_ + si];
    }
    sig[b * A_ + a] = (__bf16)v;
  }
  __syncthreads();

  const int wave = tid >> 5;
  const int lane = tid & 31;
  const int mt0  = ms * 16 + wave;                   // first M-tile
  const int mt1  = mt0 + 8;                          // second M-tile
  const int col  = lane & 15;                        // batch column
  const int koff = (lane >> 4) << 4;                 // B-operand K offset (0/16)

  v8f acc0 = {};
  v8f acc1 = {};
  const size_t abase0 = (((size_t)c * 32 + mt0) * 16) * 512 + (size_t)lane * 16;
  const size_t abase1 = (((size_t)c * 32 + mt1) * 16) * 512 + (size_t)lane * 16;
  const __bf16* A0h = Whi + abase0;
  const __bf16* A0l = Wlo + abase0;
  const __bf16* A1h = Whi + abase1;
  const __bf16* A1l = Wlo + abase1;
  const __bf16* Bp  = &sig[col * A_ + koff];

  // ---- software-pipelined K loop: prologue loads K-step 0 ----
  v16bf bc  = *(const v16bf*)(Bp);
  v16bf c0h = *(const v16bf*)(A0h);
  v16bf c0l = *(const v16bf*)(A0l);
  v16bf c1h = *(const v16bf*)(A1h);
  v16bf c1l = *(const v16bf*)(A1l);

#pragma unroll
  for (int kt = 0; kt < 15; ++kt) {
    // issue loads for K-step kt+1 (new SSA values -> distinct registers,
    // in flight while the WMMAs below run on the current fragments)
    v16bf nb  = *(const v16bf*)(Bp  + (kt + 1) * 32);
    v16bf n0h = *(const v16bf*)(A0h + (kt + 1) * 512);
    v16bf n0l = *(const v16bf*)(A0l + (kt + 1) * 512);
    v16bf n1h = *(const v16bf*)(A1h + (kt + 1) * 512);
    v16bf n1l = *(const v16bf*)(A1l + (kt + 1) * 512);

    acc0 = WMMA_BF16(c0h, bc, acc0);
    acc1 = WMMA_BF16(c1h, bc, acc1);
    acc0 = WMMA_BF16(c0l, bc, acc0);
    acc1 = WMMA_BF16(c1l, bc, acc1);

    bc = nb; c0h = n0h; c0l = n0l; c1h = n1h; c1l = n1l;
  }
  // epilogue: last K-step
  acc0 = WMMA_BF16(c0h, bc, acc0);
  acc1 = WMMA_BF16(c1h, bc, acc1);
  acc0 = WMMA_BF16(c0l, bc, acc0);
  acc1 = WMMA_BF16(c1l, bc, acc1);

  // ---- epilogue: integrate, fire, reset (lane owns 8 consecutive neurons
  //      per M-tile; C/D layout: VGPR v -> M = v + 8*(lane>=16), N = lane%16)
  const int   b    = col;
  const int   moff = (lane >> 4) << 3;
  const float th   = thresholds[c];
#pragma unroll
  for (int half = 0; half < 2; ++half) {
    const v8f& acc  = half ? acc1 : acc0;
    const int  mt   = half ? mt1 : mt0;
    const int  nbase = mt * 16 + moff;
    const size_t mbase = ((size_t)c * B_ + b) * N_ + nbase;
    const size_t bbase = (size_t)b * (C_ * N_) + (size_t)c * N_ + nbase;
#pragma unroll
    for (int v = 0; v < 8; ++v) {
      float m = memb[mbase + v] + acc[v];
      bool fired = th < m;
      buf_next[bbase + v] = fired ? (__bf16)1.0f : (__bf16)0.0f;
      memb[mbase + v]     = fired ? (m - th) : m;
    }
  }
}

// ---------------------------------------------------------------------------
// Accumulate the 1024 tapped outputs per batch into d_out.
// ---------------------------------------------------------------------------
__global__ __launch_bounds__(256) void out_gather_kernel(
    const __bf16* __restrict__ buf,
    const int* __restrict__ osc,
    const int* __restrict__ osi,
    float* __restrict__ out)
{
  int i = blockIdx.x * 256 + threadIdx.x;            // 0 .. B*OUT-1
  int b = i >> 10;
  int o = i & 1023;
  out[i] += (float)buf[(size_t)b * (C_ * N_) + osc[o] * N_ + osi[o]];
}

// ---------------------------------------------------------------------------
extern "C" void kernel_launch(void* const* d_in, const int* in_sizes, int n_in,
                              void* d_out, int out_size, void* d_ws, size_t ws_size,
                              hipStream_t stream)
{
  (void)in_sizes; (void)n_in; (void)out_size; (void)ws_size;
  const float* x         = (const float*)d_in[0];
  const float* W         = (const float*)d_in[1];
  const float* th        = (const float*)d_in[2];
  const int*   src_core  = (const int*)d_in[3];
  const int*   src_index = (const int*)d_in[4];
  const int*   osc       = (const int*)d_in[5];
  const int*   osi       = (const int*)d_in[6];
  float*       out       = (float*)d_out;

  char* ws = (char*)d_ws;
  const size_t nW     = (size_t)C_ * N_ * A_;        // 16,777,216 elements
  const size_t nState = (size_t)C_ * B_ * N_;        // 524,288 elements
  __bf16* Whi  = (__bf16*)ws;                        // 32 MB
  __bf16* Wlo  = Whi + nW;                           // 32 MB
  float*  memb = (float*)(ws + 2 * nW * sizeof(__bf16));   // 2 MB
  __bf16* buf0 = (__bf16*)(memb + nState);           // 1 MB
  __bf16* buf1 = buf0 + nState;                      // 1 MB

  prep_w_kernel<<<(unsigned)(nW / 256), 256, 0, stream>>>(W, Whi, Wlo);
  zero_state_kernel<<<(unsigned)(nState / 256), 256, 0, stream>>>(memb, buf0, buf1, out);

  for (int t = 0; t < T_; ++t) {
    __bf16* bp = (t & 1) ? buf1 : buf0;
    __bf16* bn = (t & 1) ? buf0 : buf1;
    dim3 grid(C_, 2);
    step_kernel<<<grid, 256, 0, stream>>>(x, Whi, Wlo, th, src_core, src_index,
                                          memb, bp, bn, t);
    out_gather_kernel<<<(B_ * OUT_) / 256, 256, 0, stream>>>(bn, osc, osi, out);
  }
}
